// RotorQuantLayer_48790828482957
// MI455X (gfx1250) — matrix operand order repaired and checked by reading, not compiled, not run on hardware
//
#include <hip/hip_runtime.h>
#include <stdint.h>

// ---------------------------------------------------------------------------
// RotorQuantLayer for gfx1250 (MI455X): bf16x3 WMMA GEMM (double-buffered,
// async global->LDS staging) + LDS FWHT quantizer.
//   out = FWHT(quant(FWHT(pad(xW+b))/64))/64 [:, :3072]
// ---------------------------------------------------------------------------

typedef __bf16 bf16_t;
typedef __attribute__((ext_vector_type(16))) __bf16 v16bf;
typedef __attribute__((ext_vector_type(8)))  float  v8f;

#define MDIM 8192
#define KDIM 4096
#define NDIM 3072

#if defined(__has_builtin)
#if __has_builtin(__builtin_amdgcn_global_load_async_to_lds_b128)
#define HAVE_ASYNC_COPY 1
#else
#define HAVE_ASYNC_COPY 0
#endif
#if __has_builtin(__builtin_amdgcn_s_wait_asynccnt)
#define WAIT_ASYNC0() __builtin_amdgcn_s_wait_asynccnt(0)
#else
#define WAIT_ASYNC0() asm volatile("s_wait_asynccnt 0x0" ::: "memory")
#endif
#else
#define HAVE_ASYNC_COPY 0
#define WAIT_ASYNC0()
#endif

#if HAVE_ASYNC_COPY
// Prototype (from clang diagnostic): param0 = AS(1) v4i* (global source),
// param1 = AS(3) v4i* (LDS destination), then imm offset + cpol.
typedef __attribute__((__vector_size__(16))) int v4i;
typedef __attribute__((address_space(1))) v4i glb_v4i;
typedef __attribute__((address_space(3))) v4i lds_v4i;
#endif

static __device__ __forceinline__ uint32_t pack_bf16(bf16_t a, bf16_t b) {
  union { bf16_t h[2]; uint32_t u; } p;
  p.h[0] = a; p.h[1] = b;   // little-endian: a = even K, b = odd K
  return p.u;
}

union Frag { uint4 q[2]; v16bf v; };

// ---------------------------------------------------------------------------
// Kernel 1: W [K=4096][N=3072] f32  ->  Wt_hi/Wt_lo [N][K] bf16 (hi/lo split)
// LDS-tiled 32x32 transpose so both global read and write are coalesced.
// ---------------------------------------------------------------------------
__global__ __launch_bounds__(256) void prep_w_kernel(
    const float* __restrict__ W, bf16_t* __restrict__ wt_hi,
    bf16_t* __restrict__ wt_lo) {
  __shared__ float tile[32][33];
  const int k0 = blockIdx.x * 32;
  const int n0 = blockIdx.y * 32;
  const int tx = threadIdx.x & 31;
  const int ty = threadIdx.x >> 5;   // 0..7
  for (int i = 0; i < 4; ++i) {
    const int kl = ty + 8 * i;
    tile[kl][tx] = W[(size_t)(k0 + kl) * NDIM + (n0 + tx)];
  }
  __syncthreads();
  for (int i = 0; i < 4; ++i) {
    const int nl = ty + 8 * i;
    const float w = tile[tx][nl];          // tx = local k, nl = local n
    const bf16_t hi = (bf16_t)w;
    const bf16_t lo = (bf16_t)(w - (float)hi);
    const size_t o = (size_t)(n0 + nl) * KDIM + (k0 + tx);
    wt_hi[o] = hi;
    wt_lo[o] = lo;
  }
}

// ---------------------------------------------------------------------------
// Kernel 2: bf16x3 WMMA GEMM.  y[m][n] = sum_k x[m][k]*W[k][n] + b[n]
// WG tile 64(M) x 256(N), 8 waves (wave32) as 2x4, wave tile 32x64.
// K staged in 32-wide LDS tiles, DOUBLE buffered; B tiles copied with
// GLOBAL_LOAD_ASYNC_TO_LDS_B128 (ASYNCcnt) when available, A tiles are
// f32->bf16 hi/lo split through registers. Row stride padded to 20 dwords.
// ---------------------------------------------------------------------------
__global__ __launch_bounds__(256) void gemm_bf16x3_kernel(
    const float* __restrict__ x, const bf16_t* __restrict__ wt_hi,
    const bf16_t* __restrict__ wt_lo, const float* __restrict__ bias,
    float* __restrict__ y) {
  __shared__ uint32_t AsHi[2][64 * 20];
  __shared__ uint32_t AsLo[2][64 * 20];
  __shared__ uint32_t BsHi[2][256 * 20];
  __shared__ uint32_t BsLo[2][256 * 20];

  const int tid  = threadIdx.x;
  const int lane = tid & 31;
  const int wave = tid >> 5;
  const int wm = wave >> 2;        // 0..1  (M direction)
  const int wn = wave & 3;         // 0..3  (N direction)
  const int m0 = blockIdx.x * 64;
  const int n0 = blockIdx.y * 256;

  const v8f vzero = {0.f, 0.f, 0.f, 0.f, 0.f, 0.f, 0.f, 0.f};
  v8f acc[2][4];
  for (int i = 0; i < 2; ++i)
    for (int j = 0; j < 4; ++j) acc[i][j] = vzero;

  const int l15 = lane & 15;
  // A fragment (16-bit A 16x32): lanes 0-15 hold K0-7 & K16-23,
  // lanes 16-31 hold K8-15 & K24-31, row = lane%16.
  const int aRow0 = wm * 32 + l15;
  const int aK = (lane < 16) ? 0 : 4;      // dword offset within row
  // B fragment (16-bit B 32x16): lanes 0-15 hold K0-7 & K16-23,
  // lanes 16-31 hold K8-15 & K24-31, col = lane%16.
  const int bCol0 = wn * 64 + l15;
  const int bK = (lane < 16) ? 0 : 4;      // dword offset within row

  // ---- staging helpers -----------------------------------------------------
  // A: 64x32 f32 tile -> bf16 hi/lo split, rows padded to 20 dwords.
  auto loadA = [&](int k0, float2 (&xr)[4]) {
    for (int i = 0; i < 4; ++i) {
      const int p   = tid + i * 256;   // 64 rows x 16 float2 pairs
      const int row = p >> 4;
      const int kp  = p & 15;
      xr[i] = *(const float2*)(x + (size_t)(m0 + row) * KDIM + (k0 + 2 * kp));
    }
  };
  auto storeA = [&](int bb, const float2 (&xr)[4]) {
    for (int i = 0; i < 4; ++i) {
      const int p   = tid + i * 256;
      const int row = p >> 4;
      const int kp  = p & 15;
      const bf16_t h0 = (bf16_t)xr[i].x, h1 = (bf16_t)xr[i].y;
      const bf16_t l0 = (bf16_t)(xr[i].x - (float)h0);
      const bf16_t l1 = (bf16_t)(xr[i].y - (float)h1);
      AsHi[bb][row * 20 + kp] = pack_bf16(h0, h1);
      AsLo[bb][row * 20 + kp] = pack_bf16(l0, l1);
    }
  };
#if HAVE_ASYNC_COPY
  // B: 256 cols x 32 K of pre-transposed bf16 hi/lo, async DMA to LDS.
  auto stageB = [&](int k0, int bb) {
    const uint32_t* sH = (const uint32_t*)(wt_hi + (size_t)(n0 + tid) * KDIM + k0);
    const uint32_t* sL = (const uint32_t*)(wt_lo + (size_t)(n0 + tid) * KDIM + k0);
    for (int c = 0; c < 4; ++c) {
      __builtin_amdgcn_global_load_async_to_lds_b128(
          (glb_v4i*)(sH + 4 * c), (lds_v4i*)&BsHi[bb][tid * 20 + 4 * c], 0, 0);
      __builtin_amdgcn_global_load_async_to_lds_b128(
          (glb_v4i*)(sL + 4 * c), (lds_v4i*)&BsLo[bb][tid * 20 + 4 * c], 0, 0);
    }
  };
#else
  auto loadB = [&](int k0, uint4 (&bh)[4], uint4 (&bl)[4]) {
    const uint4* sH = (const uint4*)(wt_hi + (size_t)(n0 + tid) * KDIM + k0);
    const uint4* sL = (const uint4*)(wt_lo + (size_t)(n0 + tid) * KDIM + k0);
    for (int c = 0; c < 4; ++c) { bh[c] = sH[c]; bl[c] = sL[c]; }
  };
  auto storeB = [&](int bb, const uint4 (&bh)[4], const uint4 (&bl)[4]) {
    uint4* dH = (uint4*)&BsHi[bb][tid * 20];
    uint4* dL = (uint4*)&BsLo[bb][tid * 20];
    for (int c = 0; c < 4; ++c) { dH[c] = bh[c]; dL[c] = bl[c]; }
  };
#endif

  // ---- prologue: stage tile 0 into buffer 0 --------------------------------
  {
    float2 xr[4];
    loadA(0, xr);
    storeA(0, xr);
#if HAVE_ASYNC_COPY
    stageB(0, 0);
    WAIT_ASYNC0();
#else
    uint4 bh[4], bl[4];
    loadB(0, bh, bl);
    storeB(0, bh, bl);
#endif
  }
  __syncthreads();

  // ---- main pipeline -------------------------------------------------------
  for (int kt = 0; kt < KDIM / 32; ++kt) {
    const int cur = kt & 1;
    const int nxt = cur ^ 1;
    const bool more = (kt + 1) < (KDIM / 32);
    const int k0n = (kt + 1) * 32;

#if HAVE_ASYNC_COPY
    if (more) stageB(k0n, nxt);     // ASYNCcnt DMA fills next buffer
#endif
    float2 xr[4];
    if (more) loadA(k0n, xr);       // global f32 prefetch (LOADcnt)
#if !HAVE_ASYNC_COPY
    uint4 bh[4], bl[4];
    if (more) loadB(k0n, bh, bl);
#endif

    // ---- fragment loads from current buffer
    Frag aH[2], aL[2], bHf[4], bLf[4];
    for (int im = 0; im < 2; ++im) {
      const uint4* pH = (const uint4*)&AsHi[cur][(aRow0 + im * 16) * 20 + aK];
      const uint4* pL = (const uint4*)&AsLo[cur][(aRow0 + im * 16) * 20 + aK];
      aH[im].q[0] = pH[0]; aH[im].q[1] = pH[2];   // +8 dwords = K+16
      aL[im].q[0] = pL[0]; aL[im].q[1] = pL[2];
    }
    for (int in = 0; in < 4; ++in) {
      const uint4* pH = (const uint4*)&BsHi[cur][(bCol0 + in * 16) * 20 + bK];
      const uint4* pL = (const uint4*)&BsLo[cur][(bCol0 + in * 16) * 20 + bK];
      bHf[in].q[0] = pH[0]; bHf[in].q[1] = pH[2];
      bLf[in].q[0] = pL[0]; bLf[in].q[1] = pL[2];
    }

    // ---- bf16x3: hi*hi + hi*lo + lo*hi, fp32 accumulate
    for (int im = 0; im < 2; ++im)
      for (int in = 0; in < 4; ++in) {
        acc[im][in] = __builtin_amdgcn_wmma_f32_16x16x32_bf16(
            false, aH[im].v, false, bHf[in].v, (short)0, acc[im][in], false, false);
        acc[im][in] = __builtin_amdgcn_wmma_f32_16x16x32_bf16(
            false, aH[im].v, false, bLf[in].v, (short)0, acc[im][in], false, false);
        acc[im][in] = __builtin_amdgcn_wmma_f32_16x16x32_bf16(
            false, aL[im].v, false, bHf[in].v, (short)0, acc[im][in], false, false);
      }

    // ---- finish staging next buffer, then one barrier per K-step
    if (more) {
      storeA(nxt, xr);
#if !HAVE_ASYNC_COPY
      storeB(nxt, bh, bl);
#else
      WAIT_ASYNC0();
#endif
    }
    __syncthreads();
  }

  // ---- epilogue: C/D layout -> y (only the 3072 real columns), fuse +bias
  // C/D 16x16 f32: VGPR r, lanes 0-15 = (M=r, N=lane), lanes 16-31 = (M=8+r).
  const int mAdd = (lane < 16) ? 0 : 8;
  for (int im = 0; im < 2; ++im) {
    const int mbase = m0 + wm * 32 + im * 16 + mAdd;
    for (int in = 0; in < 4; ++in) {
      const int ncol = n0 + wn * 64 + in * 16 + l15;
      const float bv = bias[ncol];
      const v8f a = acc[im][in];
      for (int r = 0; r < 8; ++r)
        y[(size_t)(mbase + r) * NDIM + ncol] = a[r] + bv;
    }
  }
}

// ---------------------------------------------------------------------------
// Kernel 3: per-row FWHT(4096) -> uniform quantize -> FWHT -> write 3072.
// Pad columns 3072..4095 are exactly zero (never materialized in y).
// ---------------------------------------------------------------------------
__global__ __launch_bounds__(256) void fwht_quant_kernel(
    const float* __restrict__ y, float* __restrict__ out) {
  __shared__ float buf[4096];
  const int t = threadIdx.x;
  const size_t row = blockIdx.x;

  for (int i = 0; i < 16; ++i) {
    const int c = t + 256 * i;
    buf[c] = (c < NDIM) ? y[row * NDIM + c] : 0.0f;
  }
  // forward FWHT
  for (int h = 1; h < 4096; h <<= 1) {
    __syncthreads();
    for (int p = t; p < 2048; p += 256) {
      const int i0 = ((p & ~(h - 1)) << 1) | (p & (h - 1));
      const int i1 = i0 + h;
      const float a = buf[i0], b = buf[i1];
      buf[i0] = a + b;
      buf[i1] = a - b;
    }
  }
  __syncthreads();
  // y_rot = buf/64; idx = clamp(rint(y_rot/0.25), -8, 7); q = idx*0.25
  for (int i = 0; i < 16; ++i) {
    const int c = t + 256 * i;
    float idx = rintf(buf[c] * (1.0f / 16.0f));   // (1/64)/0.25 = 1/16
    idx = fminf(7.0f, fmaxf(-8.0f, idx));
    buf[c] = idx * 0.25f;
  }
  // inverse (same) FWHT
  for (int h = 1; h < 4096; h <<= 1) {
    __syncthreads();
    for (int p = t; p < 2048; p += 256) {
      const int i0 = ((p & ~(h - 1)) << 1) | (p & (h - 1));
      const int i1 = i0 + h;
      const float a = buf[i0], b = buf[i1];
      buf[i0] = a + b;
      buf[i1] = a - b;
    }
  }
  __syncthreads();
  for (int i = 0; i < 12; ++i) {                  // 3072 = 12*256
    const int c = t + 256 * i;
    out[row * NDIM + c] = buf[c] * (1.0f / 64.0f);
  }
}

// ---------------------------------------------------------------------------
// Launch: prep -> GEMM -> FWHT/quant (all on `stream`, deterministic).
// Workspace layout: Wt_hi (25.17 MB) | Wt_lo (25.17 MB) | y (100.66 MB)
// ---------------------------------------------------------------------------
extern "C" void kernel_launch(void* const* d_in, const int* in_sizes, int n_in,
                              void* d_out, int out_size, void* d_ws,
                              size_t ws_size, hipStream_t stream) {
  const float* x = (const float*)d_in[0];   // (4,2048,4096) f32
  const float* W = (const float*)d_in[1];   // (4096,3072)  f32
  const float* b = (const float*)d_in[2];   // (3072,)      f32
  float* out = (float*)d_out;               // (4,2048,3072) f32

  char* ws = (char*)d_ws;
  const size_t WT_BYTES = (size_t)NDIM * KDIM * sizeof(bf16_t);  // 25165824
  bf16_t* wt_hi = (bf16_t*)(ws);
  bf16_t* wt_lo = (bf16_t*)(ws + WT_BYTES);
  float* y = (float*)(ws + 2 * WT_BYTES);   // [8192][3072] f32

  prep_w_kernel<<<dim3(KDIM / 32, NDIM / 32), 256, 0, stream>>>(W, wt_hi, wt_lo);
  gemm_bf16x3_kernel<<<dim3(MDIM / 64, NDIM / 256), 256, 0, stream>>>(
      x, wt_hi, wt_lo, b, y);
  fwht_quant_kernel<<<dim3(MDIM), 256, 0, stream>>>(y, out);
}
